// SpatialSuppressionAttention_73589969650200
// MI455X (gfx1250) — compile-verified, matrix operand order checked by reading, not run on hardware
//
#include <hip/hip_runtime.h>
#include <hip/hip_bf16.h>

#ifndef __has_builtin
#define __has_builtin(x) 0
#endif

// ---------------------------------------------------------------------------
// Types for CDNA5 WMMA (gfx1250): v_wmma_f32_16x16x32_bf16, D = A*B + C
// ---------------------------------------------------------------------------
typedef __attribute__((ext_vector_type(16))) __bf16 v16bf;
typedef __attribute__((ext_vector_type(8)))  float  v8f;

union FragBF {
    v16bf v;
    uint4 q[2];   // two 16-byte halves: elems 0..7 and 8..15
};

__device__ __forceinline__ unsigned short f2bf(float f) {
    unsigned int u = __float_as_uint(f);
    unsigned int r = u + 0x7FFFu + ((u >> 16) & 1u);   // round-to-nearest-even
    return (unsigned short)(r >> 16);
}

__device__ __forceinline__ v8f wmma_bf16(const FragBF& a, const FragBF& b, v8f c) {
    return __builtin_amdgcn_wmma_f32_16x16x32_bf16(
        false, a.v, false, b.v, (short)0, c, false, false);
}

// ---------------------------------------------------------------------------
// CDNA5 async global->LDS copy (ASYNCcnt-tracked, bypasses VGPRs).
// VDST operand = wave-relative LDS byte address; flat addresses of __shared__
// objects carry the LDS offset in their low 32 bits (ISA 10.2 aperture rules).
// ---------------------------------------------------------------------------
__device__ __forceinline__ void async_copy16(const void* g, void* lds) {
    unsigned loff = (unsigned)(unsigned long long)lds;
    asm volatile("global_load_async_to_lds_b128 %0, %1, off"
                 :: "v"(loff), "v"(g) : "memory");
}

__device__ __forceinline__ void async_wait0() {
#if __has_builtin(__builtin_amdgcn_s_wait_asynccnt)
    __builtin_amdgcn_s_wait_asynccnt(0);
#else
    asm volatile("s_wait_asynccnt 0x0" ::: "memory");
#endif
}

#define D_MODEL 768
#define NHEAD   12
#define DK      64
#define SEQ     1024
#define BATCH   8

// ---------------------------------------------------------------------------
// Kernel 1: fp32 -> bf16 conversion
// ---------------------------------------------------------------------------
__global__ void cvt_f32_bf16(const float* __restrict__ x,
                             unsigned short* __restrict__ y, int n) {
    int i = blockIdx.x * blockDim.x + threadIdx.x;
    if (i < n) y[i] = f2bf(x[i]);
}

// ---------------------------------------------------------------------------
// Kernel 2: WMMA GEMM  Y = X(bf16)[M x 768] @ W(bf16)[768 x 768]^T + bias
// WG = 4 waves. WG computes a 16(M) x 256(N) strip; wave w owns a 16x64
// sub-strip (4 accumulators) so each A fragment feeds 4 back-to-back WMMAs.
// A rows staged in LDS via async global->LDS copies.
// mode 0: -> Q bf16 [B,H,S,64], scaled by 1/sqrt(64)
// mode 1: -> K bf16 [B,H,S,64]
// mode 2: -> V bf16 transposed [B,H,64,S]
// mode 3: -> fp32 [M,768] (final output projection)
// ---------------------------------------------------------------------------
__global__ __launch_bounds__(128)
void gemm16_bf16(const unsigned short* __restrict__ X,
                 const unsigned short* __restrict__ W,
                 const float* __restrict__ bias,
                 void* __restrict__ out, int mode) {
    __shared__ uint4 As4[16 * D_MODEL / 8];   // 16 rows x 768 bf16 = 24 KB

    const int tid  = threadIdx.x;
    const int gx   = blockIdx.x;              // M-tile (16 rows)
    const int gy   = blockIdx.y;              // 256-col N group
    const int wave = tid >> 5;
    const int lane = tid & 31;
    const int lm   = lane & 15;
    const int hi8  = (lane >> 4) ? 8 : 0;     // K-half select per ISA A layout

    // Stage the 16x768 A tile (contiguous in memory) into LDS, async.
    const uint4* src = (const uint4*)(X + (size_t)gx * 16 * D_MODEL);
    for (int i = tid; i < 16 * D_MODEL / 8; i += 128)
        async_copy16(src + i, As4 + i);
    async_wait0();
    __syncthreads();

    const int nbase = gy * 256 + wave * 64;
    const unsigned short* wr0 = W + (size_t)(nbase + 0 * 16 + lm) * D_MODEL;
    const unsigned short* wr1 = W + (size_t)(nbase + 1 * 16 + lm) * D_MODEL;
    const unsigned short* wr2 = W + (size_t)(nbase + 2 * 16 + lm) * D_MODEL;
    const unsigned short* wr3 = W + (size_t)(nbase + 3 * 16 + lm) * D_MODEL;

    v8f acc0 = {}, acc1 = {}, acc2 = {}, acc3 = {};
    const int arow = lm * (D_MODEL / 8);
#pragma unroll 2
    for (int k0 = 0; k0 < D_MODEL; k0 += 32) {
        FragBF a;
        a.q[0] = As4[arow + (k0 + hi8) / 8];
        a.q[1] = As4[arow + (k0 + hi8) / 8 + 2];
        FragBF b0, b1, b2, b3;
        b0.q[0] = *(const uint4*)(wr0 + k0 + hi8);
        b0.q[1] = *(const uint4*)(wr0 + k0 + 16 + hi8);
        b1.q[0] = *(const uint4*)(wr1 + k0 + hi8);
        b1.q[1] = *(const uint4*)(wr1 + k0 + 16 + hi8);
        b2.q[0] = *(const uint4*)(wr2 + k0 + hi8);
        b2.q[1] = *(const uint4*)(wr2 + k0 + 16 + hi8);
        b3.q[0] = *(const uint4*)(wr3 + k0 + hi8);
        b3.q[1] = *(const uint4*)(wr3 + k0 + 16 + hi8);
        __builtin_prefetch(wr0 + k0 + 64, 0, 1);   // global_prefetch_b8
        __builtin_prefetch(wr2 + k0 + 64, 0, 1);
        acc0 = wmma_bf16(a, b0, acc0);
        acc1 = wmma_bf16(a, b1, acc1);
        acc2 = wmma_bf16(a, b2, acc2);
        acc3 = wmma_bf16(a, b3, acc3);
    }

    // Epilogue: C/D layout — VGPR i holds M = hi8 + i, N = lane % 16.
    v8f accs[4] = {acc0, acc1, acc2, acc3};
#pragma unroll
    for (int j = 0; j < 4; j++) {
        const int n = nbase + j * 16 + lm;
        const float bv = bias[n];
#pragma unroll
        for (int i = 0; i < 8; i++) {
            const int M = hi8 + i;
            const int r = gx * 16 + M;
            float val = accs[j][i] + bv;
            if (mode == 3) {
                ((float*)out)[(size_t)r * D_MODEL + n] = val;
            } else {
                if (mode == 0) val *= 0.125f;          // 1/sqrt(DK) folded into Q
                const int bb = r >> 10, s = r & 1023;
                const int h = n >> 6, d = n & 63;
                unsigned short* o = (unsigned short*)out;
                if (mode == 2)
                    o[(((size_t)(bb * NHEAD + h)) * DK + d) * SEQ + s] = f2bf(val);
                else
                    o[(((size_t)(bb * NHEAD + h)) * SEQ + s) * DK + d] = f2bf(val);
            }
        }
    }
}

// ---------------------------------------------------------------------------
// Kernel 3: fused scores + depthwise 3x3 conv suppression + online softmax + P@V
// One WG (4 waves / 128 threads) per (b, h, 16-query block).
// Key dim processed in 4 tiles of 256 with +/-1 halo columns and rows.
// ---------------------------------------------------------------------------
#define SC_STRIDE 292          // 288 cols + pad
#define PB_STRIDE 264          // 256 cols + pad (keeps 16B alignment: 528B/row)

__global__ __launch_bounds__(128)
void SpatialSuppressionAttention_attn(const unsigned short* __restrict__ Q,
                                      const unsigned short* __restrict__ K,
                                      const unsigned short* __restrict__ Vt,
                                      const float* __restrict__ convw,
                                      unsigned short* __restrict__ O) {
    __shared__ float sc[18 * SC_STRIDE];             // halo score tile (fp32)
    __shared__ unsigned short pbuf[16 * PB_STRIDE];  // probs tile (bf16)
    __shared__ float alp[16], lrow[16], cw[9];

    const int tid  = threadIdx.x;
    const int lane = tid & 31;
    const int wave = tid >> 5;
    const int lm   = lane & 15;
    const int hi8  = (lane >> 4) ? 8 : 0;

    const int gid = blockIdx.x;
    const int qb  = gid & 63;
    const int h   = (gid >> 6) % NHEAD;
    const int b   = gid / (64 * NHEAD);
    const int q0  = qb * 16;

    const size_t bh = (size_t)(b * NHEAD + h);
    const unsigned short* Qb = Q  + bh * SEQ * DK;
    const unsigned short* Kb = K  + bh * SEQ * DK;
    const unsigned short* Vb = Vt + bh * DK * SEQ;

    if (tid < 9) cw[tid] = convw[h * 9 + tid];
    __syncthreads();

    // Phase-B mapping: 8 lanes per query row (row br, sub-slice bs of 32 keys)
    const int br = tid >> 3, bs = tid & 7;
    float m_r = -1e30f, l_r = 0.0f;

    v8f oacc = {};                 // 16 queries x 16 d-cols per wave
    const int n0 = wave * 16;      // this wave's d-slice

    for (int kt = 0; kt < 4; kt++) {
        const int k0 = kt * 256;

        // ---- Phase A: scores for rows q0-1..q0+16, cols k0-16..k0+271 ----
        // Two shifted 16-row WMMA tiles (rt=0 -> rows q0-1.., rt=1 -> q0+1..)
        // x 18 column tiles; distributed over 4 waves.
        for (int t = wave; t < 36; t += 4) {
            const int rt = t / 18, ct = t % 18;
            int rowg = q0 - 1 + rt * 2 + lm;
            rowg = rowg < 0 ? 0 : (rowg > SEQ - 1 ? SEQ - 1 : rowg);
            int colg = k0 - 16 + ct * 16 + lm;
            colg = colg < 0 ? 0 : (colg > SEQ - 1 ? SEQ - 1 : colg);
            const unsigned short* ap = Qb + (size_t)rowg * DK;
            const unsigned short* bp = Kb + (size_t)colg * DK;
            v8f acc = {};
#pragma unroll
            for (int d0 = 0; d0 < DK; d0 += 32) {
                FragBF a, bm;
                a.q[0]  = *(const uint4*)(ap + d0 + hi8);
                a.q[1]  = *(const uint4*)(ap + d0 + 16 + hi8);
                bm.q[0] = *(const uint4*)(bp + d0 + hi8);
                bm.q[1] = *(const uint4*)(bp + d0 + 16 + hi8);
                acc = wmma_bf16(a, bm, acc);
            }
#pragma unroll
            for (int i = 0; i < 8; i++) {
                const int ldsrow = rt * 2 + hi8 + i;   // ldsrow <-> global q0-1+ldsrow
                sc[ldsrow * SC_STRIDE + ct * 16 + lm] = acc[i];
            }
        }
        __syncthreads();

        // ---- Phase B: conv suppression + online softmax ----
        float adj[32];
        float tmax = -1e30f;
#pragma unroll
        for (int kk = 0; kk < 32; kk++) {
            const int kl = bs * 32 + kk;
            const int kg = k0 + kl;
            const int cc = 16 + kl;                 // local column of kg
            const float s00 = sc[(br + 1) * SC_STRIDE + cc];
            float supp = 0.0f;
#pragma unroll
            for (int di = 0; di < 3; di++) {
                const int qg = q0 + br + di - 1;
                if (qg < 0 || qg > SEQ - 1) continue;   // SAME padding -> 0
                const int rr = br + di;
#pragma unroll
                for (int dj = 0; dj < 3; dj++) {
                    const int kg2 = kg + dj - 1;
                    if (kg2 < 0 || kg2 > SEQ - 1) continue;
                    supp += cw[di * 3 + dj] * sc[rr * SC_STRIDE + cc + dj - 1];
                }
            }
            const float a = s00 - supp;
            adj[kk] = a;
            tmax = fmaxf(tmax, a);
        }
        // reduce across the 8 lanes owning this row (wave32 shuffles)
        tmax = fmaxf(tmax, __shfl_xor(tmax, 1));
        tmax = fmaxf(tmax, __shfl_xor(tmax, 2));
        tmax = fmaxf(tmax, __shfl_xor(tmax, 4));
        const float m_new = fmaxf(m_r, tmax);
        const float alpha = __expf(m_r - m_new);
        float psum = 0.0f;
#pragma unroll
        for (int kk = 0; kk < 32; kk++) {
            const float p = __expf(adj[kk] - m_new);
            psum += p;
            pbuf[br * PB_STRIDE + bs * 32 + kk] = f2bf(p);
        }
        psum += __shfl_xor(psum, 1);
        psum += __shfl_xor(psum, 2);
        psum += __shfl_xor(psum, 4);
        l_r = l_r * alpha + psum;
        m_r = m_new;
        if (bs == 0) { alp[br] = alpha; lrow[br] = l_r; }
        __syncthreads();

        // ---- Phase C: O = O*alpha + P @ V (WMMA over 8 k-steps of 32) ----
#pragma unroll
        for (int i = 0; i < 8; i++) oacc[i] *= alp[hi8 + i];
        const unsigned short* vrow = Vb + (size_t)(n0 + lm) * SEQ;
#pragma unroll
        for (int kk0 = 0; kk0 < 256; kk0 += 32) {
            FragBF a, bm;
            const unsigned short* pr = pbuf + lm * PB_STRIDE + kk0;
            a.q[0]  = *(const uint4*)(pr + hi8);
            a.q[1]  = *(const uint4*)(pr + 16 + hi8);
            const unsigned short* vp = vrow + k0 + kk0;
            bm.q[0] = *(const uint4*)(vp + hi8);
            bm.q[1] = *(const uint4*)(vp + 16 + hi8);
            oacc = wmma_bf16(a, bm, oacc);
        }
    }
    __syncthreads();

    // ---- Epilogue: normalize by l and store bf16 [B,S,H*64] ----
#pragma unroll
    for (int i = 0; i < 8; i++) {
        const int M = hi8 + i;
        const float val = oacc[i] / lrow[M];
        const size_t row = (size_t)b * SEQ + q0 + M;
        O[row * D_MODEL + h * DK + n0 + lm] = f2bf(val);
    }
}

// ---------------------------------------------------------------------------
// Host-side orchestration (all on `stream`, graph-capture safe).
// ---------------------------------------------------------------------------
extern "C" void kernel_launch(void* const* d_in, const int* in_sizes, int n_in,
                              void* d_out, int out_size, void* d_ws, size_t ws_size,
                              hipStream_t stream) {
    (void)in_sizes; (void)n_in; (void)out_size; (void)ws_size;
    const float* q     = (const float*)d_in[0];
    const float* k     = (const float*)d_in[1];
    const float* v     = (const float*)d_in[2];
    const float* wq    = (const float*)d_in[3];
    const float* bq    = (const float*)d_in[4];
    const float* wk    = (const float*)d_in[5];
    const float* bk    = (const float*)d_in[6];
    const float* wv    = (const float*)d_in[7];
    const float* bv    = (const float*)d_in[8];
    const float* wo    = (const float*)d_in[9];
    const float* bo    = (const float*)d_in[10];
    const float* convw = (const float*)d_in[11];

    char* ws = (char*)d_ws;
    size_t off = 0;
    auto alloc = [&](size_t bytes) -> char* {
        char* p = ws + off;
        off += (bytes + 255) & ~(size_t)255;
        return p;
    };

    const size_t MD = (size_t)BATCH * SEQ * D_MODEL;   // 6,291,456
    const size_t WD = (size_t)D_MODEL * D_MODEL;       //   589,824

    unsigned short* Xq  = (unsigned short*)alloc(MD * 2);
    unsigned short* Xk  = (unsigned short*)alloc(MD * 2);
    unsigned short* Xv  = (unsigned short*)alloc(MD * 2);
    unsigned short* Wqb = (unsigned short*)alloc(WD * 2);
    unsigned short* Wkb = (unsigned short*)alloc(WD * 2);
    unsigned short* Wvb = (unsigned short*)alloc(WD * 2);
    unsigned short* Wob = (unsigned short*)alloc(WD * 2);
    unsigned short* Qbf = (unsigned short*)alloc(MD * 2);   // [B,H,S,64]
    unsigned short* Kbf = (unsigned short*)alloc(MD * 2);   // [B,H,S,64]
    unsigned short* Vtb = (unsigned short*)alloc(MD * 2);   // [B,H,64,S]
    unsigned short* AO  = (unsigned short*)alloc(MD * 2);   // attn out [B,S,768]

    // 1) fp32 -> bf16
    const int nMD = (int)MD, nWD = (int)WD;
    cvt_f32_bf16<<<(nMD + 255) / 256, 256, 0, stream>>>(q,  Xq,  nMD);
    cvt_f32_bf16<<<(nMD + 255) / 256, 256, 0, stream>>>(k,  Xk,  nMD);
    cvt_f32_bf16<<<(nMD + 255) / 256, 256, 0, stream>>>(v,  Xv,  nMD);
    cvt_f32_bf16<<<(nWD + 255) / 256, 256, 0, stream>>>(wq, Wqb, nWD);
    cvt_f32_bf16<<<(nWD + 255) / 256, 256, 0, stream>>>(wk, Wkb, nWD);
    cvt_f32_bf16<<<(nWD + 255) / 256, 256, 0, stream>>>(wv, Wvb, nWD);
    cvt_f32_bf16<<<(nWD + 255) / 256, 256, 0, stream>>>(wo, Wob, nWD);

    // 2) projections (WMMA GEMMs); each wave computes a 16x64 strip
    dim3 gg(BATCH * SEQ / 16, D_MODEL / 256), gb(128);
    gemm16_bf16<<<gg, gb, 0, stream>>>(Xq, Wqb, bq, Qbf, 0);
    gemm16_bf16<<<gg, gb, 0, stream>>>(Xk, Wkb, bk, Kbf, 1);
    gemm16_bf16<<<gg, gb, 0, stream>>>(Xv, Wvb, bv, Vtb, 2);

    // 3) fused scores + conv suppression + softmax + P@V
    SpatialSuppressionAttention_attn<<<BATCH * NHEAD * (SEQ / 16), 128, 0, stream>>>(
        Qbf, Kbf, Vtb, convw, AO);

    // 4) output projection -> fp32 d_out
    gemm16_bf16<<<gg, gb, 0, stream>>>(AO, Wob, bo, d_out, 3);
}
